// GAT_83597243450115
// MI455X (gfx1250) — matrix operand order repaired and checked by reading, not compiled, not run on hardware
//
#include <hip/hip_runtime.h>
#include <math.h>

typedef __bf16 bf16;
typedef __attribute__((ext_vector_type(16))) __bf16 v16bf;
typedef __attribute__((ext_vector_type(8)))  __bf16 v8bf;
typedef __attribute__((ext_vector_type(8)))  float  v8f;

#define BB 8
#define NN 2048
#define DD 512

// D = A(16x32 bf16) * B(32x16 bf16) + C(16x16 f32)
__device__ __forceinline__ v8f wmma_bf16(v16bf a, v16bf b, v8f c) {
    return __builtin_amdgcn_wmma_f32_16x16x32_bf16(
        false, a, false, b, (short)0, c, false, false);
}

// A-fragment (16x32, 16-bit): lane<16 holds K = 0..15 ; lane>=16 holds K = 16..31
// caller passes p = rowptr + ks + half*16  (16-byte aligned)
__device__ __forceinline__ v16bf frag_a(const bf16* p) {
    v8bf lo = *(const v8bf*)p;        // K: base .. base+7
    v8bf hi = *(const v8bf*)(p + 8);  // K: base+8 .. base+15
    v16bf r;
#pragma unroll
    for (int i = 0; i < 8; ++i) { r[i] = lo[i]; r[i + 8] = hi[i]; }
    return r;
}

__device__ __forceinline__ float wave_max(float v) {
#pragma unroll
    for (int m = 16; m > 0; m >>= 1) v = fmaxf(v, __shfl_xor(v, m, 32));
    return v;
}
__device__ __forceinline__ float wave_sum(float v) {
#pragma unroll
    for (int m = 16; m > 0; m >>= 1) v += __shfl_xor(v, m, 32);
    return v;
}

// ---------------------------------------------------------------------------
// f32 -> bf16 conversion (grid-stride)
__global__ void gat_cvt_bf16(const float* __restrict__ in, bf16* __restrict__ out, long n) {
    long i = (long)blockIdx.x * blockDim.x + threadIdx.x;
    long s = (long)gridDim.x * blockDim.x;
    for (; i < n; i += s) out[i] = (bf16)in[i];
}

// Pack W[512][512] into B-fragment layout:
// Wp[((kb*32 + nt)*32 + lane)*16 + e] ; n = nt*16 + lane%16, k = kb*32 + (lane>>4)*16 + e
__global__ void gat_pack_w(const float* __restrict__ W, bf16* __restrict__ Wp) {
    int idx = blockIdx.x * 256 + threadIdx.x;      // 262144 total
    int e    = idx & 15;
    int lane = (idx >> 4) & 31;
    int nt   = (idx >> 9) & 31;
    int kb   = idx >> 14;
    int n = nt * 16 + (lane & 15);
    int k = kb * 32 + (lane >> 4) * 16 + e;
    Wp[idx] = (bf16)W[(size_t)k * DD + n];
}

// ---------------------------------------------------------------------------
// Y[M,512] = X[M,512] @ W + bias ; tmode=0: row-major bf16 (Q,K); tmode=1: Yt[b][d][N] (V)
__global__ void gat_proj(const bf16* __restrict__ X, const bf16* __restrict__ Wp,
                         const float* __restrict__ bias, bf16* __restrict__ Y, int tmode) {
    const int tid = threadIdx.x;
    const int wv = tid >> 5, lane = tid & 31;
    const int laneN = lane & 15, half = lane >> 4;
    const int m0 = blockIdx.x * 16;
    const int nt = blockIdx.y * 8 + wv;
    const int n0 = nt * 16;

    const bf16* Arow = X + (size_t)(m0 + laneN) * DD;
    v8f acc = {};
#pragma unroll
    for (int kb = 0; kb < DD / 32; ++kb) {
        v16bf a = frag_a(Arow + kb * 32 + half * 16);
        v16bf b = *(const v16bf*)(Wp + (((size_t)kb * 32 + nt) * 32 + lane) * 16);
        acc = wmma_bf16(a, b, acc);
    }
    float bv = bias[n0 + laneN];
#pragma unroll
    for (int i = 0; i < 8; ++i) acc[i] += bv;

    if (tmode == 0) {
#pragma unroll
        for (int r = 0; r < 8; ++r)
            Y[(size_t)(m0 + half * 8 + r) * DD + n0 + laneN] = (bf16)acc[r];
    } else {
        int bidx = m0 >> 11;
        int mm0  = (m0 & (NN - 1)) + half * 8;
        v8bf pk;
#pragma unroll
        for (int r = 0; r < 8; ++r) pk[r] = (bf16)acc[r];
        *(v8bf*)(Y + ((size_t)bidx * DD + n0 + laneN) * NN + mm0) = pk;
    }
}

// ---------------------------------------------------------------------------
// Flash attention: grid (N/32, B), block 256 (8 waves). 32 queries per workgroup.
// Q,K row-major bf16 [B*N, D]; Vt transposed bf16 [B][D][N]; out f32 [B,N,D]
__global__ void __launch_bounds__(256)
gat_attn(const bf16* __restrict__ Q, const bf16* __restrict__ K,
         const bf16* __restrict__ Vt, float* __restrict__ out) {
    __shared__ bf16  sQ[32 * DD];       // 32 KB: query tile (32 rows)
    __shared__ float sS[32 * 128];      // 16 KB: raw scores for one key block
    __shared__ bf16  sP[32 * 128];      // 8 KB : probabilities (A operand of P·V)
    __shared__ float sM[32], sL[32], sC[32];

    const int tid = threadIdx.x;
    const int wv = tid >> 5, lane = tid & 31;
    const int laneN = lane & 15, half = lane >> 4;
    const int q0 = blockIdx.x * 32;
    const int b  = blockIdx.y;

    const bf16* Qb = Q + ((size_t)b * NN + q0) * DD;        // 32 contiguous rows
    const bf16* Kb = K + (size_t)b * NN * DD;
    const bf16* Vb = Vt + (size_t)b * DD * NN;

    // stage Q tile into LDS (contiguous 32*512 bf16)
    for (int i = tid; i < 32 * DD / 8; i += 256)
        ((v8bf*)sQ)[i] = ((const v8bf*)Qb)[i];
    if (tid < 32) { sM[tid] = -INFINITY; sL[tid] = 0.0f; }
    __syncthreads();

    v8f acc[2][4];
#pragma unroll
    for (int qs = 0; qs < 2; ++qs)
#pragma unroll
        for (int t = 0; t < 4; ++t) acc[qs][t] = (v8f){};

    const float invS = 0.04419417382415922f;  // 1/sqrt(512)

    for (int kb0 = 0; kb0 < NN; kb0 += 128) {
        // ---- phase 1: S(32q x 16k) per wave; K fragment reused across both q subtiles
        v8f s0 = {}, s1 = {};
        const bf16* Krow = Kb + (size_t)(kb0 + wv * 16 + laneN) * DD;
        const bf16* Q0 = sQ + laneN * DD;
        const bf16* Q1 = sQ + (16 + laneN) * DD;
        if (kb0 + 128 < NN) {   // prefetch next key block (this wave's key rows)
            __builtin_prefetch(Krow + 128 * DD, 0, 1);
            __builtin_prefetch(Vb + (size_t)(wv * 64 + laneN) * NN + kb0 + 128, 0, 1);
        }
#pragma unroll
        for (int ks = 0; ks < DD; ks += 32) {
            v16bf kf = *(const v16bf*)(Krow + ks + half * 16);
            v16bf a0 = frag_a(Q0 + ks + half * 16);
            v16bf a1 = frag_a(Q1 + ks + half * 16);
            s0 = wmma_bf16(a0, kf, s0);
            s1 = wmma_bf16(a1, kf, s1);
        }
#pragma unroll
        for (int r = 0; r < 8; ++r) {
            sS[(half * 8 + r) * 128 + wv * 16 + laneN]        = s0[r];
            sS[(16 + half * 8 + r) * 128 + wv * 16 + laneN]   = s1[r];
        }
        __syncthreads();

        // ---- phase 2: online softmax, wave-parallel: 8 waves x 4 rows, 4 cols/lane
#pragma unroll
        for (int rr = 0; rr < 4; ++rr) {
            const int r = wv * 4 + rr;
            float x[4];
#pragma unroll
            for (int i = 0; i < 4; ++i) x[i] = sS[r * 128 + lane * 4 + i] * invS;
            float mx = fmaxf(fmaxf(x[0], x[1]), fmaxf(x[2], x[3]));
            mx = wave_max(mx);
            const float mo = sM[r];
            const float mn = fmaxf(mo, mx);
            float sum = 0.0f;
#pragma unroll
            for (int i = 0; i < 4; ++i) {
                float p = __expf(x[i] - mn);
                sum += p;
                sP[r * 128 + lane * 4 + i] = (bf16)p;
            }
            sum = wave_sum(sum);
            if (lane == 0) {
                float corr = __expf(mo - mn);
                sM[r] = mn; sL[r] = sL[r] * corr + sum; sC[r] = corr;
            }
        }
        __syncthreads();

        // ---- phase 3: rescale accumulators, O += P * V (wave owns 64 d-cols)
#pragma unroll
        for (int qs = 0; qs < 2; ++qs) {
            float corr[8];
#pragma unroll
            for (int r = 0; r < 8; ++r) corr[r] = sC[qs * 16 + half * 8 + r];
#pragma unroll
            for (int t = 0; t < 4; ++t)
#pragma unroll
                for (int r = 0; r < 8; ++r) acc[qs][t][r] *= corr[r];
        }
#pragma unroll
        for (int kc = 0; kc < 128; kc += 32) {
            v16bf p0 = frag_a(sP + laneN * 128 + kc + half * 16);
            v16bf p1 = frag_a(sP + (16 + laneN) * 128 + kc + half * 16);
#pragma unroll
            for (int t = 0; t < 4; ++t) {
                const int d = wv * 64 + t * 16 + laneN;
                v16bf vf = *(const v16bf*)(Vb + (size_t)d * NN + kb0 + kc + half * 16);
                acc[0][t] = wmma_bf16(p0, vf, acc[0][t]);
                acc[1][t] = wmma_bf16(p1, vf, acc[1][t]);
            }
        }
        __syncthreads();
    }

    // ---- epilogue: divide by l, store f32 row-major ----
#pragma unroll
    for (int qs = 0; qs < 2; ++qs) {
        float rinv[8];
#pragma unroll
        for (int r = 0; r < 8; ++r) rinv[r] = 1.0f / sL[qs * 16 + half * 8 + r];
#pragma unroll
        for (int t = 0; t < 4; ++t) {
            const int d = wv * 64 + t * 16 + laneN;
#pragma unroll
            for (int r = 0; r < 8; ++r)
                out[((size_t)b * NN + q0 + qs * 16 + half * 8 + r) * DD + d] =
                    acc[qs][t][r] * rinv[r];
        }
    }
}

// ---------------------------------------------------------------------------
extern "C" void kernel_launch(void* const* d_in, const int* in_sizes, int n_in,
                              void* d_out, int out_size, void* d_ws, size_t ws_size,
                              hipStream_t stream) {
    (void)in_sizes; (void)n_in; (void)out_size; (void)ws_size;
    const float* x    = (const float*)d_in[0];
    const float* Wq_w = (const float*)d_in[1];
    const float* Wq_b = (const float*)d_in[2];
    const float* Wk_w = (const float*)d_in[3];
    const float* Wk_b = (const float*)d_in[4];
    const float* Wv_w = (const float*)d_in[5];
    const float* Wv_b = (const float*)d_in[6];
    float* out = (float*)d_out;

    const size_t M = (size_t)BB * NN;             // 16384
    char* w = (char*)d_ws;
    bf16* xb  = (bf16*)w;                          w += M * DD * sizeof(bf16);        // 16 MB
    bf16* Wpq = (bf16*)w;                          w += (size_t)DD * DD * sizeof(bf16);
    bf16* Wpk = (bf16*)w;                          w += (size_t)DD * DD * sizeof(bf16);
    bf16* Wpv = (bf16*)w;                          w += (size_t)DD * DD * sizeof(bf16);
    bf16* Qb  = (bf16*)w;                          w += M * DD * sizeof(bf16);        // 16 MB
    bf16* Kb  = (bf16*)w;                          w += M * DD * sizeof(bf16);        // 16 MB
    bf16* Vt  = (bf16*)w;                          // [B][D][N], 16 MB

    gat_cvt_bf16<<<4096, 256, 0, stream>>>(x, xb, (long)M * DD);
    gat_pack_w<<<1024, 256, 0, stream>>>(Wq_w, Wpq);
    gat_pack_w<<<1024, 256, 0, stream>>>(Wk_w, Wpk);
    gat_pack_w<<<1024, 256, 0, stream>>>(Wv_w, Wpv);

    dim3 gp(M / 16, (DD / 16) / 8);               // 1024 x 4
    gat_proj<<<gp, 256, 0, stream>>>(xb, Wpq, Wq_b, Qb, 0);
    gat_proj<<<gp, 256, 0, stream>>>(xb, Wpk, Wk_b, Kb, 0);
    gat_proj<<<gp, 256, 0, stream>>>(xb, Wpv, Wv_b, Vt, 1);

    dim3 ga(NN / 32, BB);                         // 64 x 8
    gat_attn<<<ga, 256, 0, stream>>>(Qb, Kb, Vt, out);
}
